// SimpleVDforVQA_50508815401590
// MI455X (gfx1250) — matrix operand order repaired and checked by reading, not compiled, not run on hardware
//
#include <hip/hip_runtime.h>
#include <hip/hip_bf16.h>
#include <math.h>

// ---------------------------------------------------------------------------
// MI455X (gfx1250) implementation.
// GEMMs run on v_wmma_f32_16x16x32_bf16 (wave32, 16x16 tiles, K=32/instr).
// WMMA operands are pre-packed fragment-major (512 bf16 per 16x32/32x16 tile,
// 32 contiguous bytes per lane). A tiles stream through ping-pong LDS buffers
// (register-prefetched a chunk ahead); B fragments are register double-
// buffered so loads of k-step kt+1 overlap the wmmas of k-step kt. VQ argmin
// is fused into the distance-GEMM epilogue.
// ---------------------------------------------------------------------------

typedef __attribute__((ext_vector_type(16))) __bf16 v16bf;
typedef __attribute__((ext_vector_type(8)))  float  v8f;
typedef unsigned short ushort_t;

#define B_      64
#define CIN     2048
#define COUT    768
#define KCODES  2048
#define HH      16
#define WW      16
#define NTOK    (B_ * HH * WW)      // 16384
#define PFD     (COUT / 2)          // 384
#define KT1     (CIN / 32)          // 64  k-tiles, conv GEMM
#define OT      (COUT / 16)         // 48  o-tiles, conv GEMM
#define KT2     (COUT / 32)         // 24  k-tiles, VQ GEMM
#define JT      (KCODES / 16)       // 128 code-tiles, VQ GEMM

// f32 -> bf16 round-to-nearest-even (bit trick)
__device__ __forceinline__ ushort_t f2bf(float f) {
  unsigned int u = __float_as_uint(f);
  u = u + 0x7FFFu + ((u >> 16) & 1u);
  return (ushort_t)(u >> 16);
}

// A-fragment (16x32 bf16, MxK) element -> (lane, elem) per CDNA5 ISA 7.12.2
__device__ __forceinline__ int a_lane(int m, int k) {
  return m + (((k >> 3) & 1) << 4);
}
__device__ __forceinline__ int a_elem(int k) {
  int base = (k & 16) ? 4 : 0;           // VGPR 4..7 hold K=16..31 half
  int v = base + (((k & 15) & 7) >> 1);  // pair index within half
  return 2 * v + (k & 1);
}
// inverse: lane half (kb = lane>=16 ? 8 : 0) + elem e -> k
__device__ __forceinline__ int a_k_from_elem(int kb, int e) {
  int v = e >> 1;
  return ((v >= 4) ? 16 : 0) + kb + ((v & 3) << 1) + (e & 1);
}
// B-fragment (32x16 bf16, KxN): lane = n + ((k>>3)&1)*16, elem = (k&7) + (k>=16)*8

__device__ __forceinline__ v8f wmma_bf16(v16bf a, v16bf b, v8f c) {
  return __builtin_amdgcn_wmma_f32_16x16x32_bf16(false, a, false, b,
                                                 (short)0, c, false, false);
}

// ---------------------------------------------------------------------------
// K0a: pack conv_w [COUT,CIN] -> B-fragments Wp, tile grid [kt][ot]
__global__ void pack_w_kernel(const float* __restrict__ conv_w,
                              ushort_t* __restrict__ Wp) {
  size_t id = (size_t)blockIdx.x * 256 + threadIdx.x;  // KT1*OT*512
  int e = id & 15;
  int lane = (id >> 4) & 31;
  size_t tile = id >> 9;
  int ot = (int)(tile % OT);
  int kt = (int)(tile / OT);
  int n = lane & 15;
  int k = (e & 7) + ((lane & 16) >> 1) + ((e & 8) << 1);
  int o = ot * 16 + n;
  int c = kt * 32 + k;
  Wp[tile * 512 + lane * 16 + e] = f2bf(conv_w[(size_t)o * CIN + c]);
}

// K0b: pack codebook [KCODES,COUT] -> B-fragments Cp, tile grid [kt][jt]
__global__ void pack_cb_kernel(const float* __restrict__ codebook,
                               ushort_t* __restrict__ Cp) {
  size_t id = (size_t)blockIdx.x * 256 + threadIdx.x;
  int e = id & 15;
  int lane = (id >> 4) & 31;
  size_t tile = id >> 9;
  int jt = (int)(tile % JT);
  int kt = (int)(tile / JT);
  int n = lane & 15;
  int k = (e & 7) + ((lane & 16) >> 1) + ((e & 8) << 1);
  int j = jt * 16 + n;
  int c = kt * 32 + k;
  Cp[tile * 512 + lane * 16 + e] = f2bf(codebook[(size_t)j * COUT + c]);
}

// K0c: codebook row squared norms
__global__ void cb_norms_kernel(const float* __restrict__ codebook,
                                float* __restrict__ cbn) {
  int j = blockIdx.x;
  float s = 0.f;
  for (int c = threadIdx.x; c < COUT; c += 256) {
    float v = codebook[(size_t)j * COUT + c];
    s += v * v;
  }
  __shared__ float sm[256];
  sm[threadIdx.x] = s;
  __syncthreads();
  for (int k = 128; k > 0; k >>= 1) {
    if (threadIdx.x < (unsigned)k) sm[threadIdx.x] += sm[threadIdx.x + k];
    __syncthreads();
  }
  if (threadIdx.x == 0) cbn[j] = sm[0];
}

// K0d: sine positional embedding, stored [t][c] (t = y*16+x)
__global__ void pos_kernel(const int* __restrict__ pad_h, const int* __restrict__ pad_w,
                           const int* __restrict__ imh, const int* __restrict__ imw,
                           float* __restrict__ pos) {
  int c = blockIdx.x;      // 0..767
  int t = threadIdx.x;     // 0..255
  int y = t >> 4, x = t & 15;
  int PH = *pad_h, PW = *pad_w, IH = *imh, IW = *imw;
  const float eps = 1e-6f;
  const float TWO_PI = 6.28318530717958647692f;
  float ycum = 0.f, ytot = 0.f, xcum = 0.f, xtot = 0.f;
  int xs_x = (int)floorf((float)x * ((float)PW / (float)WW));
  int ys_y = (int)floorf((float)y * ((float)PH / (float)HH));
  for (int i = 0; i < HH; ++i) {
    int ys_i = (int)floorf((float)i * ((float)PH / (float)HH));
    float m = (ys_i < IH && xs_x < IW) ? 1.f : 0.f;
    if (i <= y) ycum += m;
    ytot += m;
  }
  for (int i = 0; i < WW; ++i) {
    int xs_i = (int)floorf((float)i * ((float)PW / (float)WW));
    float m = (ys_y < IH && xs_i < IW) ? 1.f : 0.f;
    if (i <= x) xcum += m;
    xtot += m;
  }
  float yn = ycum / (ytot + eps) * TWO_PI;
  float xn = xcum / (xtot + eps) * TWO_PI;
  float e = (c < PFD) ? yn : xn;
  int jf = (c < PFD) ? c : (c - PFD);
  float dim_t = powf(10000.f, 2.f * floorf((float)jf * 0.5f) / (float)PFD);
  float val = e / dim_t;
  pos[(size_t)t * COUT + c] = (jf & 1) ? cosf(val) : sinf(val);
}

// K0e: visibility mask -> int output tail
__global__ void vmask_kernel(const int* __restrict__ pad_h, const int* __restrict__ pad_w,
                             const int* __restrict__ imh, const int* __restrict__ imw,
                             int* __restrict__ vmask) {
  int t = threadIdx.x;
  int b = blockIdx.x;
  int y = t >> 4, x = t & 15;
  int ys = (int)floorf((float)y * ((float)(*pad_h) / (float)HH));
  int xs = (int)floorf((float)x * ((float)(*pad_w) / (float)WW));
  vmask[(size_t)b * 256 + t] = (ys < *imh && xs < *imw) ? 1 : 0;
}

// ---------------------------------------------------------------------------
// K1: 2x2 maxpool + bf16 + pack into A-fragments A1p.
// One thread produces one full 32-byte lane fragment (16 bf16); adjacent
// lanes map to adjacent x -> coalesced f2 loads, vector b128 stores.
__global__ void maxpool_pack_kernel(const float* __restrict__ img,
                                    ushort_t* __restrict__ A1p) {
  size_t id = (size_t)blockIdx.x * 256 + threadIdx.x;  // NTOK/16 * KT1 * 32
  int lane = (int)(id & 31);
  size_t tile = id >> 5;                 // mt*KT1 + kt
  int kt = (int)(tile % KT1);
  int mt = (int)(tile / KT1);
  int m = lane & 15;
  int kb = (lane & 16) ? 8 : 0;
  int n = mt * 16 + m;                   // token
  int b = n >> 8;
  int t = n & 255;
  int y = t >> 4, x = t & 15;
  v16bf frag;
#pragma unroll
  for (int e = 0; e < 16; ++e) {
    int k = a_k_from_elem(kb, e);
    int c = kt * 32 + k;
    const float* p = img + ((((size_t)b * CIN + c) * 32 + (size_t)(2 * y)) * 32 + (size_t)(2 * x));
    float2 r0 = *(const float2*)p;
    float2 r1 = *(const float2*)(p + 32);
    float mx = fmaxf(fmaxf(r0.x, r0.y), fmaxf(r1.x, r1.y));
    ((ushort_t*)&frag)[e] = f2bf(mx);
  }
  *(v16bf*)(A1p + tile * 512 + (size_t)lane * 16) = frag;
}

// ---------------------------------------------------------------------------
// K2: conv GEMM  xq[n,o] = A1 @ conv_w^T + bias.  grid = NTOK/16 blocks,
// 8 waves; wave w owns o-tiles [6w..6w+5].  A tiles stream through ping-pong
// LDS buffers with register prefetch one chunk ahead; B fragments register
// double-buffered across k-steps.
#define ACH 16                       // k-tiles per chunk (16 KiB per buffer)
#define NCH (KT1 / ACH)              // 4 chunks
#define CHU4 (ACH * 512 / 8)         // 1024 uint4 per chunk
__global__ void conv_gemm_kernel(const ushort_t* __restrict__ A1p,
                                 const ushort_t* __restrict__ Wp,
                                 const float* __restrict__ bias,
                                 ushort_t* __restrict__ A2p,
                                 float* __restrict__ rown) {
  const int mt = blockIdx.x;
  const int lane = threadIdx.x & 31;
  const int wave = threadIdx.x >> 5;

  __shared__ ushort_t As[2][ACH * 512];  // 2 x 16 KiB ping-pong
  __shared__ float smem[8][16];

  v8f acc[6];
#pragma unroll
  for (int j = 0; j < 6; ++j)
#pragma unroll
    for (int r = 0; r < 8; ++r) acc[j][r] = 0.f;

  const ushort_t* Bbase = Wp + (size_t)(wave * 6) * 512 + (size_t)lane * 16;
  v16bf b_cur[6], b_nxt[6];
#pragma unroll
  for (int j = 0; j < 6; ++j)
    b_cur[j] = *(const v16bf*)(Bbase + (size_t)j * 512);

  const uint4* Asrc = (const uint4*)(A1p + (size_t)mt * KT1 * 512);
  uint4 pf[4];
  // chunk 0: load + store (no compute to overlap yet)
#pragma unroll
  for (int j = 0; j < 4; ++j) pf[j] = Asrc[threadIdx.x + j * 256];
#pragma unroll
  for (int j = 0; j < 4; ++j) ((uint4*)As[0])[threadIdx.x + j * 256] = pf[j];
  __syncthreads();

  int cur = 0;
  for (int c = 0; c < NCH; ++c) {
    if (c + 1 < NCH) {  // prefetch next chunk into registers; compute hides it
#pragma unroll
      for (int j = 0; j < 4; ++j)
        pf[j] = Asrc[(size_t)(c + 1) * CHU4 + threadIdx.x + j * 256];
    }
#pragma unroll 4
    for (int kt = 0; kt < ACH; ++kt) {
      int gk = c * ACH + kt;
      int gn = (gk + 1 < KT1) ? gk + 1 : gk;
#pragma unroll
      for (int j = 0; j < 6; ++j)
        b_nxt[j] = *(const v16bf*)(Bbase + ((size_t)gn * OT + j) * 512);
      const v16bf a = *(const v16bf*)(&As[cur][kt * 512 + lane * 16]);
#pragma unroll
      for (int j = 0; j < 6; ++j) acc[j] = wmma_bf16(a, b_cur[j], acc[j]);
#pragma unroll
      for (int j = 0; j < 6; ++j) b_cur[j] = b_nxt[j];
    }
    if (c + 1 < NCH) {
#pragma unroll
      for (int j = 0; j < 4; ++j)
        ((uint4*)As[cur ^ 1])[threadIdx.x + j * 256] = pf[j];
    }
    __syncthreads();
    cur ^= 1;
  }

  // epilogue: +bias, write bf16 A2p fragments, accumulate row sumsq
  const int hi = lane >> 4;
  const int col = lane & 15;
  float rsq[8];
#pragma unroll
  for (int r = 0; r < 8; ++r) rsq[r] = 0.f;

#pragma unroll
  for (int j = 0; j < 6; ++j) {
    int ot = wave * 6 + j;
    int o = ot * 16 + col;
    float bs = bias[o];
    int kt2 = o >> 5, kk = o & 31;
#pragma unroll
    for (int r = 0; r < 8; ++r) {
      float v = acc[j][r] + bs;
      rsq[r] += v * v;
      int m = r + hi * 8;
      A2p[((size_t)mt * KT2 + kt2) * 512 + a_lane(m, kk) * 16 + a_elem(kk)] = f2bf(v);
    }
  }

#pragma unroll
  for (int s = 8; s >= 1; s >>= 1)
#pragma unroll
    for (int r = 0; r < 8; ++r) rsq[r] += __shfl_xor(rsq[r], s, 32);

  if (col == 0) {
#pragma unroll
    for (int r = 0; r < 8; ++r) smem[wave][hi * 8 + r] = rsq[r];
  }
  __syncthreads();
  if (threadIdx.x < 16) {
    float s = 0.f;
#pragma unroll
    for (int w = 0; w < 8; ++w) s += smem[w][threadIdx.x];
    rown[(size_t)mt * 16 + threadIdx.x] = s;
  }
}

// ---------------------------------------------------------------------------
// K3: VQ distance GEMM + fused argmin.  d = |inp|^2 - 2 inp.cb + |cb|^2.
// Two token tiles per block (halves codebook L2 traffic, 32 FLOP/B from L2);
// all 48 A-tiles staged in LDS once; each wave owns 16 code-tiles, processed
// in groups of 2 with B double-buffering.
__global__ void vq_argmin_kernel(const ushort_t* __restrict__ A2p,
                                 const ushort_t* __restrict__ Cp,
                                 const float* __restrict__ rown,
                                 const float* __restrict__ cbn,
                                 int* __restrict__ idxout) {
  const int mt0 = blockIdx.x * 2;
  const int lane = threadIdx.x & 31;
  const int wave = threadIdx.x >> 5;
  const int hi = lane >> 4;
  const int col = lane & 15;

  __shared__ ushort_t As[2 * KT2 * 512];  // 48 KiB, [mi][kt][512]
  __shared__ float sv[8][16];
  __shared__ int si[8][16];

  {  // one contiguous stage: tiles of mt0 and mt0+1 are adjacent in A2p
    const uint4* src = (const uint4*)(A2p + (size_t)mt0 * KT2 * 512);
    uint4* dst = (uint4*)As;
    for (int i = threadIdx.x; i < 2 * KT2 * 512 / 8; i += 256) dst[i] = src[i];
  }
  __syncthreads();

  float rn[2][8];
#pragma unroll
  for (int mi = 0; mi < 2; ++mi)
#pragma unroll
    for (int r = 0; r < 8; ++r)
      rn[mi][r] = rown[(size_t)(mt0 + mi) * 16 + r + hi * 8];

  float minv[2][8];
  int mini[2][8];
#pragma unroll
  for (int mi = 0; mi < 2; ++mi)
#pragma unroll
    for (int r = 0; r < 8; ++r) { minv[mi][r] = __builtin_inff(); mini[mi][r] = 0; }

  for (int g = 0; g < 8; ++g) {
    const int jt0 = wave * 16 + g * 2;
    const ushort_t* Bbase = Cp + (size_t)jt0 * 512 + (size_t)lane * 16;
    v8f acc[2][2];
#pragma unroll
    for (int mi = 0; mi < 2; ++mi)
#pragma unroll
      for (int q = 0; q < 2; ++q)
#pragma unroll
        for (int r = 0; r < 8; ++r) acc[mi][q][r] = 0.f;

    v16bf b_cur[2], b_nxt[2];
#pragma unroll
    for (int q = 0; q < 2; ++q)
      b_cur[q] = *(const v16bf*)(Bbase + (size_t)q * 512);

#pragma unroll 4
    for (int kt = 0; kt < KT2; ++kt) {
      int kn = (kt + 1 < KT2) ? kt + 1 : kt;
#pragma unroll
      for (int q = 0; q < 2; ++q)
        b_nxt[q] = *(const v16bf*)(Bbase + ((size_t)kn * JT + q) * 512);
      const v16bf a0 = *(const v16bf*)(As + (0 * KT2 + kt) * 512 + lane * 16);
      const v16bf a1 = *(const v16bf*)(As + (1 * KT2 + kt) * 512 + lane * 16);
#pragma unroll
      for (int q = 0; q < 2; ++q) acc[0][q] = wmma_bf16(a0, b_cur[q], acc[0][q]);
#pragma unroll
      for (int q = 0; q < 2; ++q) acc[1][q] = wmma_bf16(a1, b_cur[q], acc[1][q]);
#pragma unroll
      for (int q = 0; q < 2; ++q) b_cur[q] = b_nxt[q];
    }

#pragma unroll
    for (int q = 0; q < 2; ++q) {
      int j = (jt0 + q) * 16 + col;
      float cn = cbn[j];
#pragma unroll
      for (int mi = 0; mi < 2; ++mi)
#pragma unroll
        for (int r = 0; r < 8; ++r) {
          float d = rn[mi][r] - 2.f * acc[mi][q][r] + cn;
          if (d < minv[mi][r]) { minv[mi][r] = d; mini[mi][r] = j; }
        }
    }
  }

  // cross-lane (16-lane group) min with smallest-index tie-break
#pragma unroll
  for (int s = 8; s >= 1; s >>= 1) {
#pragma unroll
    for (int mi = 0; mi < 2; ++mi)
#pragma unroll
      for (int r = 0; r < 8; ++r) {
        float ov = __shfl_xor(minv[mi][r], s, 32);
        int oi = __shfl_xor(mini[mi][r], s, 32);
        if (ov < minv[mi][r] || (ov == minv[mi][r] && oi < mini[mi][r])) {
          minv[mi][r] = ov;
          mini[mi][r] = oi;
        }
      }
  }

#pragma unroll
  for (int mi = 0; mi < 2; ++mi) {
    __syncthreads();
    if (col == 0) {
#pragma unroll
      for (int r = 0; r < 8; ++r) { sv[wave][hi * 8 + r] = minv[mi][r]; si[wave][hi * 8 + r] = mini[mi][r]; }
    }
    __syncthreads();
    if (threadIdx.x < 16) {
      float bv = sv[0][threadIdx.x];
      int bi = si[0][threadIdx.x];
#pragma unroll
      for (int w = 1; w < 8; ++w) {
        float v = sv[w][threadIdx.x];
        int ii = si[w][threadIdx.x];
        if (v < bv || (v == bv && ii < bi)) { bv = v; bi = ii; }
      }
      idxout[(size_t)(mt0 + mi) * 16 + threadIdx.x] = bi;
    }
  }
}

// ---------------------------------------------------------------------------
// K4: gather codebook row + pos add + LayerNorm.  One block per token.
__global__ void out_ln_kernel(const float* __restrict__ codebook,
                              const int* __restrict__ idx,
                              const float* __restrict__ pos,
                              const float* __restrict__ ln_g,
                              const float* __restrict__ ln_b,
                              float* __restrict__ out) {
  int n = blockIdx.x;         // 0..NTOK-1
  int t = n & 255;
  int code = idx[n];
  float v[3], s = 0.f, s2 = 0.f;
#pragma unroll
  for (int i = 0; i < 3; ++i) {
    int c = threadIdx.x + i * 256;
    float val = codebook[(size_t)code * COUT + c] + pos[(size_t)t * COUT + c];
    v[i] = val;
    s += val;
    s2 += val * val;
  }
  __shared__ float sm[256], sm2[256];
  sm[threadIdx.x] = s;
  sm2[threadIdx.x] = s2;
  __syncthreads();
  for (int k = 128; k > 0; k >>= 1) {
    if (threadIdx.x < (unsigned)k) {
      sm[threadIdx.x] += sm[threadIdx.x + k];
      sm2[threadIdx.x] += sm2[threadIdx.x + k];
    }
    __syncthreads();
  }
  float mu = sm[0] * (1.f / COUT);
  float var = sm2[0] * (1.f / COUT) - mu * mu;
  float inv = rsqrtf(var + 1e-5f);
#pragma unroll
  for (int i = 0; i < 3; ++i) {
    int c = threadIdx.x + i * 256;
    out[(size_t)n * COUT + c] = (v[i] - mu) * inv * ln_g[c] + ln_b[c];
  }
}

// ---------------------------------------------------------------------------
extern "C" void kernel_launch(void* const* d_in, const int* in_sizes, int n_in,
                              void* d_out, int out_size, void* d_ws, size_t ws_size,
                              hipStream_t stream) {
  (void)in_sizes; (void)n_in; (void)out_size; (void)ws_size;
  const float* img      = (const float*)d_in[0];
  const float* conv_w   = (const float*)d_in[1];
  const float* conv_b   = (const float*)d_in[2];
  const float* codebook = (const float*)d_in[3];
  const float* ln_g     = (const float*)d_in[4];
  const float* ln_b     = (const float*)d_in[5];
  const int* pad_h      = (const int*)d_in[6];
  const int* pad_w      = (const int*)d_in[7];
  const int* imh        = (const int*)d_in[8];
  const int* imw        = (const int*)d_in[9];

  float* out = (float*)d_out;
  int* vmask = (int*)((char*)d_out + (size_t)NTOK * COUT * 4);

  // workspace carve-up (all naturally aligned)
  char* w = (char*)d_ws;
  ushort_t* A1p = (ushort_t*)w; w += (size_t)NTOK * CIN * 2;      // 64 MiB
  ushort_t* Wp  = (ushort_t*)w; w += (size_t)CIN * COUT * 2;      //  3 MiB
  ushort_t* Cp  = (ushort_t*)w; w += (size_t)COUT * KCODES * 2;   //  3 MiB
  ushort_t* A2p = (ushort_t*)w; w += (size_t)NTOK * COUT * 2;     // 24 MiB
  float* rown   = (float*)w;    w += (size_t)NTOK * 4;
  float* cbn    = (float*)w;    w += (size_t)KCODES * 4;
  float* pos    = (float*)w;    w += (size_t)256 * COUT * 4;
  int* idx      = (int*)w;      w += (size_t)NTOK * 4;

  // prep / packing
  pack_w_kernel<<<KT1 * OT * 2, 256, 0, stream>>>(conv_w, Wp);
  pack_cb_kernel<<<KT2 * JT * 2, 256, 0, stream>>>(codebook, Cp);
  cb_norms_kernel<<<KCODES, 256, 0, stream>>>(codebook, cbn);
  pos_kernel<<<COUT, 256, 0, stream>>>(pad_h, pad_w, imh, imw, pos);
  vmask_kernel<<<B_, 256, 0, stream>>>(pad_h, pad_w, imh, imw, vmask);

  // maxpool + pack A1 (one thread = one 32B lane fragment)
  maxpool_pack_kernel<<<(NTOK / 16) * KT1 * 32 / 256, 256, 0, stream>>>(img, A1p);

  // conv GEMM (WMMA bf16) -> A2p, rown
  conv_gemm_kernel<<<NTOK / 16, 256, 0, stream>>>(A1p, Wp, conv_b, A2p, rown);

  // VQ GEMM (WMMA bf16) + fused argmin, 32 tokens per block
  vq_argmin_kernel<<<NTOK / 32, 256, 0, stream>>>(A2p, Cp, rown, cbn, idx);

  // gather + pos + LayerNorm
  out_ln_kernel<<<NTOK, 256, 0, stream>>>(codebook, idx, pos, ln_g, ln_b, out);
}
